// Non_local_15513421873702
// MI455X (gfx1250) — compile-verified, matrix operand order checked by reading, not run on hardware
//
#include <hip/hip_runtime.h>
#include <hip/hip_bf16.h>

// Non-local attention block for MI455X (gfx1250).
// B=8, C=256, N=H*W=3072, D=128.  ~45 GFLOP total vs ~50MB HBM traffic
// -> matrix-core bound.  All GEMMs run on v_wmma_f32_16x16x32_bf16.
// Flash-style fusion: the NxN attention matrix never hits memory.
// LDS staging uses TDM (tensor_load_to_lds) and async global->LDS copies
// where the toolchain exposes them; guarded fallbacks otherwise.

#define BATCH 8
#define CCH   256
#define NTOK  3072
#define DDIM  128
#define INV_N (1.0f / 3072.0f)

typedef __bf16 bf16;
typedef __attribute__((ext_vector_type(16))) __bf16 v16bf;
typedef __attribute__((ext_vector_type(8)))  __bf16 v8bf;
typedef __attribute__((ext_vector_type(8)))  float  v8f;
typedef __attribute__((ext_vector_type(4)))  unsigned int u32x4;
typedef __attribute__((ext_vector_type(8)))  int i32x8;
typedef __attribute__((ext_vector_type(4)))  int i32x4;

// Pointer-to-AS(1)/AS(3) 16-byte granules for the async LDS copy builtin.
typedef __attribute__((address_space(1))) i32x4* gptr_v4i;
typedef __attribute__((address_space(3))) i32x4* lptr_v4i;

#if __has_builtin(__builtin_amdgcn_global_load_async_to_lds_b128)
#define HAS_ASYNC 1
#else
#define HAS_ASYNC 0
#endif

#if __has_builtin(__builtin_amdgcn_tensor_load_to_lds)
#define HAS_TDM 1
#else
#define HAS_TDM 0
#endif

union frag16 { v16bf v; v8bf h[2]; };

// Load a 16x32 bf16 A-fragment (or B-fragment of a transposed row-major
// matrix) from a row-major source: two 16B loads per lane.
__device__ inline v16bf load_frag(const bf16* p) {
  frag16 f;
  f.h[0] = *(const v8bf*)p;
  f.h[1] = *(const v8bf*)(p + 16);
  return f.v;
}

__device__ inline void wait_async0() {
#if __has_builtin(__builtin_amdgcn_s_wait_asynccnt)
  __builtin_amdgcn_s_wait_asynccnt(0);
#else
  asm volatile("s_wait_asynccnt 0" ::: "memory");
#endif
}

#if HAS_TDM
// LDS byte offset of a __shared__ object (generic -> AS(3) -> int).
__device__ inline unsigned lds_off(const void* p) {
  return (unsigned)(unsigned long long)
      (__attribute__((address_space(3))) const void*)p;
}

// 1-D contiguous TDM copy: `elems` bf16 elements global -> LDS.
// D# per CDNA5 ISA: count=1, type=2 ("image"),
// data_size=1 (2B), tile_dim0=elems (<=65535), tensor_dim0=stride=elems.
__device__ inline void tdm_load_1d(const void* gsrc, unsigned lds_addr,
                                   unsigned elems) {
  unsigned long long ga = (unsigned long long)gsrc;
  u32x4 g0;
  g0[0] = 1u;                                    // count=1, user mode
  g0[1] = lds_addr;                              // lds_addr
  g0[2] = (unsigned)ga;                          // global_addr[31:0]
  g0[3] = (unsigned)((ga >> 32) & 0x01FFFFFFu) | (2u << 30);  // type=2
  i32x8 g1;
  g1[0] = (int)(1u << 16);                       // data_size=1 (2 bytes)
  g1[1] = (int)((elems & 0xFFFFu) << 16);        // tensor_dim0[15:0]
  g1[2] = (int)(((elems >> 16) & 0xFFFFu) | (1u << 16));  // dim0 hi, dim1=1
  g1[3] = (int)((elems & 0xFFFFu) << 16);        // tile_dim0
  g1[4] = 1;                                     // tile_dim1=1
  g1[5] = (int)elems;                            // tensor_dim0_stride lo
  g1[6] = 0;
  g1[7] = 0;
  i32x4 z4 = {0, 0, 0, 0};
#if __has_include(<hip/amd_detail/amd_gfx1250_TDM.h>)
  i32x8 z8 = {0, 0, 0, 0, 0, 0, 0, 0};
  __builtin_amdgcn_tensor_load_to_lds(g0, g1, z4, z4, z8, 0);  // clang-23 form
#else
  __builtin_amdgcn_tensor_load_to_lds(g0, g1, z4, z4, 0);      // ROCm 7.2 form
#endif
}
#endif  // HAS_TDM

// Stage a contiguous bf16 block (elems multiple of 1024) into LDS.
__device__ inline void stage_block(const bf16* gsrc, bf16* ldst, int tid,
                                   int elems) {
#if HAS_ASYNC
  for (int i = tid * 8; i < elems; i += 128 * 8)
    __builtin_amdgcn_global_load_async_to_lds_b128(
        (gptr_v4i)(void*)(gsrc + i), (lptr_v4i)(void*)(ldst + i), 0, 0);
#else
  for (int i = tid; i < elems; i += 128) ldst[i] = gsrc[i];
#endif
}

// ---------------------------------------------------------------------------
// Kernel 0: convert the four weight matrices f32 -> bf16 (each 128*256 elems)
// ---------------------------------------------------------------------------
__global__ void conv_w_kernel(const float* a, const float* b, const float* c,
                              const float* d, bf16* A, bf16* B, bf16* C,
                              bf16* D, int n) {
  int i = blockIdx.x * blockDim.x + threadIdx.x;
  if (i < n) {
    A[i] = (bf16)a[i];
    B[i] = (bf16)b[i];
    C[i] = (bf16)c[i];
    D[i] = (bf16)d[i];
  }
}

// ---------------------------------------------------------------------------
// Kernel 1: fused projections.  out[b][n][d] = sum_c x[b][c][n]*w[d][c] + bias
// for theta, phi, g.  Block = 128 threads (4 waves), 64 n-rows per block.
// x tile (64 x 256) staged transposed into LDS once, reused for 24 d-tiles.
// ---------------------------------------------------------------------------
__global__ __launch_bounds__(128) void proj_kernel(
    const float* __restrict__ x,
    const bf16* __restrict__ wTh, const bf16* __restrict__ wPh,
    const bf16* __restrict__ wGm,
    const float* __restrict__ bTh, const float* __restrict__ bPh,
    const float* __restrict__ bGm,
    bf16* __restrict__ th, bf16* __restrict__ ph, bf16* __restrict__ gg) {
  __shared__ bf16 lsX[64 * CCH];  // [m][c], 32 KB

  const int b  = blockIdx.x / (NTOK / 64);
  const int n0 = (blockIdx.x % (NTOK / 64)) * 64;
  const int tid = threadIdx.x, wave = tid >> 5, lane = tid & 31;
  const int cb = (lane >> 4) * 8;

  // stage x transposed: lsX[m][c] = x[b][c][n0+m]; coalesced over m
  for (int i = tid; i < 64 * CCH; i += 128) {
    int c = i >> 6, m = i & 63;
    lsX[m * CCH + c] = (bf16)x[((size_t)b * CCH + c) * NTOK + n0 + m];
  }
  __syncthreads();

  // preload 8 A-fragments covering K = C = 256 for this wave's 16 rows
  v16bf aX[8];
  {
    const int row = wave * 16 + (lane & 15);
    for (int kf = 0; kf < 8; ++kf)
      aX[kf] = load_frag(&lsX[row * CCH + kf * 32 + cb]);
  }

  for (int ot = 0; ot < 24; ++ot) {  // 0-7 theta, 8-15 phi, 16-23 g
    const bf16*  wsel = (ot < 8) ? wTh : (ot < 16) ? wPh : wGm;
    const float* bsel = (ot < 8) ? bTh : (ot < 16) ? bPh : bGm;
    bf16*        osel = (ot < 8) ? th  : (ot < 16) ? ph  : gg;
    const int d0 = (ot & 7) * 16;

    v8f acc = {};
    const bf16* wrow = wsel + (size_t)(d0 + (lane & 15)) * CCH + cb;
    for (int kf = 0; kf < 8; ++kf) {
      v16bf bW = load_frag(wrow + kf * 32);
      acc = __builtin_amdgcn_wmma_f32_16x16x32_bf16(
          false, aX[kf], false, bW, (short)0, acc, false, false);
    }
    const float bias = bsel[d0 + (lane & 15)];
    bf16* orow = osel + ((size_t)b * NTOK + n0 + wave * 16) * DDIM +
                 d0 + (lane & 15);
    for (int v = 0; v < 8; ++v) {
      int r = v + 8 * (lane >> 4);
      orow[(size_t)r * DDIM] = (bf16)(acc[v] + bias);
    }
  }
}

// ---------------------------------------------------------------------------
// Kernel 2: fused attention  Y = (Theta Phi^T / N) G  without materializing f.
// Block = 128 threads (4 waves), 64 Theta rows; loop over 48 chunks of 64
// Phi/G rows.  Theta staged via TDM, Phi via async global->LDS copies.
// ---------------------------------------------------------------------------
__global__ __launch_bounds__(128) void attn_kernel(
    const bf16* __restrict__ th, const bf16* __restrict__ ph,
    const bf16* __restrict__ gg, bf16* __restrict__ yout) {
  __shared__ bf16 lsT[64 * DDIM];       // Theta block    16 KB
  __shared__ bf16 lsP[64 * DDIM];       // Phi chunk      16 KB
  __shared__ bf16 lsG[DDIM * 64];       // G chunk, T     16 KB
  __shared__ bf16 lsS[4][16 * 64];      // per-wave S      8 KB

  const int b  = blockIdx.x / (NTOK / 64);
  const int n0 = (blockIdx.x % (NTOK / 64)) * 64;
  const int tid = threadIdx.x, wave = tid >> 5, lane = tid & 31;
  const int cb = (lane >> 4) * 8;

  const bf16* thg = th + ((size_t)b * NTOK + n0) * DDIM;
#if HAS_TDM
  if (wave == 0) {  // one TDM descriptor moves the whole 16KB Theta block
    tdm_load_1d(thg, lds_off(lsT), 64 * DDIM);
    __builtin_amdgcn_s_wait_tensorcnt(0);
  }
#else
  for (int i = tid; i < 64 * DDIM; i += 128) lsT[i] = thg[i];
#endif
  __syncthreads();

  v16bf aT[4];  // Theta A-fragments, K = D = 128
  {
    const int row = wave * 16 + (lane & 15);
    for (int kf = 0; kf < 4; ++kf)
      aT[kf] = load_frag(&lsT[row * DDIM + kf * 32 + cb]);
  }

  v8f acc[8];
  for (int i = 0; i < 8; ++i) acc[i] = (v8f){};

  for (int m0 = 0; m0 < NTOK; m0 += 64) {
    __syncthreads();  // previous chunk's LDS reads complete
    const bf16* pG = ph + ((size_t)b * NTOK + m0) * DDIM;
    const bf16* gG = gg + ((size_t)b * NTOK + m0) * DDIM;
    stage_block(pG, lsP, tid, 64 * DDIM);        // async path when available
    for (int i = tid; i < 64 * DDIM; i += 128) {
      int m = i >> 7, d = i & 127;
      lsG[d * 64 + m] = gG[i];  // transpose so B-fragments load contiguously
    }
    if (m0 + 64 < NTOK)  // prefetch next chunk into GL2
      __builtin_prefetch(pG + 64 * DDIM, 0, 1);
#if HAS_ASYNC
    wait_async0();
#endif
    __syncthreads();

    // S (16 x 64) = Theta_wave (16 x 128) * Phi_chunk^T
    for (int t = 0; t < 4; ++t) {
      v8f s = {};
      for (int kf = 0; kf < 4; ++kf) {
        v16bf bP = load_frag(&lsP[(t * 16 + (lane & 15)) * DDIM + kf * 32 + cb]);
        s = __builtin_amdgcn_wmma_f32_16x16x32_bf16(
            false, aT[kf], false, bP, (short)0, s, false, false);
      }
      // C-layout -> row-major bf16 S in per-wave LDS, scaled by 1/N
      for (int v = 0; v < 8; ++v)
        lsS[wave][(v + 8 * (lane >> 4)) * 64 + t * 16 + (lane & 15)] =
            (bf16)(s[v] * INV_N);
    }
    // cross-lane exchange through LDS within the wave: wait for DS stores
    asm volatile("s_wait_dscnt 0" ::: "memory");

    // Y (16 x 128) += S (16 x 64) * G_chunk (64 x 128)
    for (int kf = 0; kf < 2; ++kf) {
      v16bf aS = load_frag(&lsS[wave][(lane & 15) * 64 + kf * 32 + cb]);
      for (int dt = 0; dt < 8; ++dt) {
        v16bf bG = load_frag(&lsG[(dt * 16 + (lane & 15)) * 64 + kf * 32 + cb]);
        acc[dt] = __builtin_amdgcn_wmma_f32_16x16x32_bf16(
            false, aS, false, bG, (short)0, acc[dt], false, false);
      }
    }
  }

  bf16* yrow = yout + ((size_t)b * NTOK + n0 + wave * 16) * DDIM;
  for (int dt = 0; dt < 8; ++dt)
    for (int v = 0; v < 8; ++v)
      yrow[(size_t)(v + 8 * (lane >> 4)) * DDIM + dt * 16 + (lane & 15)] =
          (bf16)acc[dt][v];
}

// ---------------------------------------------------------------------------
// Kernel 3: out[b][c][n] = BN( w_w[c,:]·y[b,n,:] + w_b[c] ) + x[b][c][n]
// One 16(n) x 16(c) WMMA tile per wave; BN folded into scale+offset.
// Result tile bounces through LDS so the residual load and final store are
// contiguous float4 accesses instead of stride-N scatters.
// ---------------------------------------------------------------------------
__global__ __launch_bounds__(128) void out_kernel(
    const bf16* __restrict__ yws, const bf16* __restrict__ ww,
    const float* __restrict__ wb, const float* __restrict__ gamma,
    const float* __restrict__ beta, const float* __restrict__ mean,
    const float* __restrict__ var, const float* __restrict__ x,
    float* __restrict__ out) {
  __shared__ float lsO[4][16][17];  // [wave][n][c], padded vs bank conflicts

  const int gtile = blockIdx.x * 4 + (threadIdx.x >> 5);
  const int tilesPerBatch = (NTOK / 16) * (CCH / 16);  // 3072
  const int b = gtile / tilesPerBatch;
  const int t = gtile % tilesPerBatch;
  const int n0 = (t / (CCH / 16)) * 16;
  const int c0 = (t % (CCH / 16)) * 16;
  const int wave = threadIdx.x >> 5, lane = threadIdx.x & 31;
  const int cb = (lane >> 4) * 8;

  v8f acc = {};
  const bf16* yrow = yws + ((size_t)b * NTOK + n0 + (lane & 15)) * DDIM + cb;
  const bf16* wrow = ww + (size_t)(c0 + (lane & 15)) * DDIM + cb;
  for (int kf = 0; kf < 4; ++kf) {
    v16bf aY = load_frag(yrow + kf * 32);
    v16bf bW = load_frag(wrow + kf * 32);
    acc = __builtin_amdgcn_wmma_f32_16x16x32_bf16(
        false, aY, false, bW, (short)0, acc, false, false);
  }

  // fold BN at the producer lane (col c = c0 + lane&15), park tile in LDS
  {
    const int c = c0 + (lane & 15);
    const float inv = gamma[c] * __frsqrt_rn(var[c] + 1e-5f);
    const float add = (wb[c] - mean[c]) * inv + beta[c];
    for (int v = 0; v < 8; ++v)
      lsO[wave][v + 8 * (lane >> 4)][lane & 15] = acc[v] * inv + add;
  }
  asm volatile("s_wait_dscnt 0" ::: "memory");

  // consumer mapping: lane -> (c row, 8-wide n segment); contiguous float4 IO
  const int cr = lane >> 1;
  const int ns = (lane & 1) * 8;
  const size_t base = ((size_t)b * CCH + c0 + cr) * NTOK + n0 + ns;
  const float4* xr = (const float4*)(x + base);
  float4* orow = (float4*)(out + base);
  float4 o0, o1;
  o0.x = lsO[wave][ns + 0][cr];
  o0.y = lsO[wave][ns + 1][cr];
  o0.z = lsO[wave][ns + 2][cr];
  o0.w = lsO[wave][ns + 3][cr];
  o1.x = lsO[wave][ns + 4][cr];
  o1.y = lsO[wave][ns + 5][cr];
  o1.z = lsO[wave][ns + 6][cr];
  o1.w = lsO[wave][ns + 7][cr];
  float4 x0 = xr[0], x1 = xr[1];
  o0.x += x0.x; o0.y += x0.y; o0.z += x0.z; o0.w += x0.w;
  o1.x += x1.x; o1.y += x1.y; o1.z += x1.z; o1.w += x1.w;
  orow[0] = o0;
  orow[1] = o1;
}

// ---------------------------------------------------------------------------
extern "C" void kernel_launch(void* const* d_in, const int* in_sizes, int n_in,
                              void* d_out, int out_size, void* d_ws,
                              size_t ws_size, hipStream_t stream) {
  const float* x       = (const float*)d_in[0];
  const float* g_w     = (const float*)d_in[1];
  const float* g_b     = (const float*)d_in[2];
  const float* theta_w = (const float*)d_in[3];
  const float* theta_b = (const float*)d_in[4];
  const float* phi_w   = (const float*)d_in[5];
  const float* phi_b   = (const float*)d_in[6];
  const float* w_w     = (const float*)d_in[7];
  const float* w_b     = (const float*)d_in[8];
  const float* bn_g    = (const float*)d_in[9];
  const float* bn_b    = (const float*)d_in[10];
  const float* bn_m    = (const float*)d_in[11];
  const float* bn_v    = (const float*)d_in[12];
  float* out = (float*)d_out;

  const size_t wElems = (size_t)DDIM * CCH;            // 32768
  const size_t pElems = (size_t)BATCH * NTOK * DDIM;   // 3,145,728

  char* ws = (char*)d_ws;
  bf16* wTh = (bf16*)ws; ws += wElems * sizeof(bf16);
  bf16* wPh = (bf16*)ws; ws += wElems * sizeof(bf16);
  bf16* wGm = (bf16*)ws; ws += wElems * sizeof(bf16);
  bf16* wWo = (bf16*)ws; ws += wElems * sizeof(bf16);
  bf16* thx = (bf16*)ws; ws += pElems * sizeof(bf16);
  bf16* phx = (bf16*)ws; ws += pElems * sizeof(bf16);
  bf16* gx  = (bf16*)ws; ws += pElems * sizeof(bf16);
  bf16* yws = (bf16*)ws; ws += pElems * sizeof(bf16);
  (void)ws_size; (void)in_sizes; (void)n_in; (void)out_size;

  conv_w_kernel<<<(int)((wElems + 255) / 256), 256, 0, stream>>>(
      theta_w, phi_w, g_w, w_w, wTh, wPh, wGm, wWo, (int)wElems);

  proj_kernel<<<BATCH * (NTOK / 64), 128, 0, stream>>>(
      x, wTh, wPh, wGm, theta_b, phi_b, g_b, thx, phx, gx);

  attn_kernel<<<BATCH * (NTOK / 64), 128, 0, stream>>>(thx, phx, gx, yws);

  out_kernel<<<BATCH * (NTOK / 16) * (CCH / 16) / 4, 128, 0, stream>>>(
      yws, wWo, w_b, bn_g, bn_b, bn_m, bn_v, x, out);
}
